// MultiFeatureExtractor_38946763440354
// MI455X (gfx1250) — matrix operand order repaired and checked by reading, not compiled, not run on hardware
//
#include <hip/hip_runtime.h>
#include <hip/hip_bf16.h>
#include <stdint.h>

// ---------------------------------------------------------------------------
// MI455X (gfx1250) implementation.
#define N_ROI 512
#define M_REF 2048
#define FEATD 1024
#define NGRP  16
#define DGRP  64
#define EMBD  64

typedef __bf16 bf16;
typedef __attribute__((ext_vector_type(16))) __bf16 v16bf;
typedef __attribute__((ext_vector_type(8)))  __bf16 v8bf;
typedef __attribute__((ext_vector_type(8)))  float  v8f;

union ABFrag { v16bf v; v8bf h[2]; };

// ---------------------------------------------------------------------------
// fp32 -> bf16 conversion (bandwidth trivial)
__global__ __launch_bounds__(256) void cvt_bf16_kernel(const float* __restrict__ in,
                                                       bf16* __restrict__ out, long long n) {
    long long i = (long long)blockIdx.x * blockDim.x + threadIdx.x;
    if (i < n) out[i] = (bf16)in[i];
}

// ---------------------------------------------------------------------------
// Generic TN GEMM:  C[z] = ( A[z] (MxK, K-fast) * B[z] (NxK, K-fast)^T + bias ) * alpha
// One wave -> 16 x (NT*16) output tile, NT WMMAs per K-step sharing the A frag.
// Block = 128 threads = 4 waves covering 16 rows x (4*NT*16) cols.
// Launch guarantees N % (4*NT*16) == 0 and M % 16 == 0: no guards, straight-line
// inner loop. Double-buffered fragments: loads for k+32 are issued before the
// WMMAs on k, so the loadcnt waits ahead of the WMMAs are partial and the next
// iteration's loads stay in flight while the matrix pipe runs.
template <int NT, bool BF16_OUT>
__global__ __launch_bounds__(128)
void gemm_tn_kernel(const bf16* __restrict__ A, const bf16* __restrict__ B,
                    void* __restrict__ Cv,
                    int K, int lda, int ldb, int ldc,
                    long long sA, long long sB, long long sC,
                    const float* __restrict__ bias, long long sBias, float alpha) {
    const int lane = threadIdx.x & 31;
    const int wave = threadIdx.x >> 5;
    const int z    = blockIdx.z;
    A += (long long)z * sA;
    B += (long long)z * sB;

    const int m0     = blockIdx.y * 16;
    const int n_base = (blockIdx.x * 4 + wave) * (NT * 16);

    const int l15 = lane & 15;   // A row within tile / B column / C column
    const int hlf = lane >> 4;   // K-half selector per ISA bf16 layouts

    // A fragment: lanes L and L+16 both hold row M=l15; K chunks:
    //   hlf==0 -> {k..k+7, k+16..k+23}, hlf==1 -> {k+8..k+15, k+24..k+31}
    const bf16* aptr = A + (long long)(m0 + l15) * lda + hlf * 8;
    // B fragment: lane = hlf*16 + n holds K = {hlf*8..hlf*8+7, 16+hlf*8..16+hlf*8+7} of column n
    const bf16* bptr[NT];
#pragma unroll
    for (int t = 0; t < NT; ++t)
        bptr[t] = B + (long long)(n_base + t * 16 + l15) * ldb + hlf * 8;

    auto load_frags = [&](int k, ABFrag& af, ABFrag (&bf)[NT]) {
        af.h[0] = *(const v8bf*)(aptr + k);
        af.h[1] = *(const v8bf*)(aptr + k + 16);
#pragma unroll
        for (int t = 0; t < NT; ++t) {
            bf[t].h[0] = *(const v8bf*)(bptr[t] + k);
            bf[t].h[1] = *(const v8bf*)(bptr[t] + k + 16);
        }
    };

    v8f acc[NT];
#pragma unroll
    for (int t = 0; t < NT; ++t) acc[t] = v8f{};

    ABFrag a0, b0[NT];
    load_frags(0, a0, b0);
    int k = 0;
#pragma unroll 2
    for (; k + 32 < K; k += 32) {
        ABFrag a1, b1[NT];
        load_frags(k + 32, a1, b1);   // prefetch next K-slice
#pragma unroll
        for (int t = 0; t < NT; ++t)
            acc[t] = __builtin_amdgcn_wmma_f32_16x16x32_bf16(
                false, a0.v, false, b0[t].v, (short)0, acc[t], false, false);
        a0 = a1;
#pragma unroll
        for (int t = 0; t < NT; ++t) b0[t] = b1[t];
    }
#pragma unroll
    for (int t = 0; t < NT; ++t)
        acc[t] = __builtin_amdgcn_wmma_f32_16x16x32_bf16(
            false, a0.v, false, b0[t].v, (short)0, acc[t], false, false);

    // Epilogue. C VGPR r: lanes 0-15 -> row r, lanes 16-31 -> row r+8; col = l15.
    const long long cbase = (long long)z * sC;
#pragma unroll
    for (int t = 0; t < NT; ++t) {
        const int ccol = n_base + t * 16 + l15;
        const float bv = bias ? bias[(long long)z * sBias + ccol] : 0.0f;
#pragma unroll
        for (int r = 0; r < 8; ++r) {
            const int crow = m0 + r + hlf * 8;
            const float v = (acc[t][r] + bv) * alpha;
            const long long cidx = cbase + (long long)crow * ldc + ccol;
            if (BF16_OUT) ((bf16*)Cv)[cidx] = (bf16)v;
            else          ((float*)Cv)[cidx] = v;
        }
    }
}

// ---------------------------------------------------------------------------
// aff_c[g][m] = sum_d u[g,d] * K[m, g*64+d]
__global__ __launch_bounds__(256)
void affc_kernel(const bf16* __restrict__ Kb, const float* __restrict__ u,
                 float* __restrict__ affc) {
    int idx = blockIdx.x * blockDim.x + threadIdx.x;
    if (idx >= NGRP * M_REF) return;
    int g = idx >> 11, m = idx & (M_REF - 1);
    const bf16*  kp = Kb + (long long)m * FEATD + g * DGRP;
    const float* up = u + g * EMBD;  // u is (GROUP,1,EMB), EMB==DG==64
    float s = 0.0f;
#pragma unroll
    for (int d = 0; d < DGRP; ++d) s += (float)kp[d] * up[d];
    affc[idx] = s;
}

// ---------------------------------------------------------------------------
// Fused position-embedding + Wg projection + log(relu(.)+1e-6) accumulation:
//   logits[g][n][m] += log(relu(Wg_b[g] + sum_e Wg[g,e]*emb_e(n,m)) + 1e-6) + 1e-12
__global__ __launch_bounds__(256)
void posw_kernel(const float* __restrict__ bbox, const float* __restrict__ bbox_ref,
                 const float* __restrict__ Wg_w, const float* __restrict__ Wg_b,
                 float* __restrict__ logits) {
    __shared__ float sWg[NGRP * EMBD];
    __shared__ float sWb[NGRP];
    for (int i = threadIdx.x; i < NGRP * EMBD; i += blockDim.x) sWg[i] = Wg_w[i];
    if (threadIdx.x < NGRP) sWb[threadIdx.x] = Wg_b[threadIdx.x];
    __syncthreads();

    const int m = blockIdx.x * blockDim.x + threadIdx.x;
    const int n = blockIdx.y;
    if (m >= M_REF) return;

    const float xmin = bbox[n * 4 + 0], ymin = bbox[n * 4 + 1];
    const float xmax = bbox[n * 4 + 2], ymax = bbox[n * 4 + 3];
    const float w  = xmax - xmin + 1.0f, h  = ymax - ymin + 1.0f;
    const float cx = 0.5f * (xmin + xmax), cy = 0.5f * (ymin + ymax);

    const float xr0 = bbox_ref[m * 4 + 0], yr0 = bbox_ref[m * 4 + 1];
    const float xr1 = bbox_ref[m * 4 + 2], yr1 = bbox_ref[m * 4 + 3];
    const float wr  = xr1 - xr0 + 1.0f, hr  = yr1 - yr0 + 1.0f;
    const float cxr = 0.5f * (xr0 + xr1), cyr = 0.5f * (yr0 + yr1);

    float p[4];
    p[0] = __logf(fabsf((cx - cxr) / w) + 0.001f);
    p[1] = __logf(fabsf((cy - cyr) / h) + 0.001f);
    p[2] = __logf(w / wr);
    p[3] = __logf(h / hr);

    // dim_mat[j] = 1000^(j/8)
    const float dimj[8] = {1.0f, 2.3713737f, 5.6234132f, 13.335214f,
                           31.622776f, 74.989418f, 177.82794f, 421.69650f};
    float accg[NGRP];
#pragma unroll
    for (int g = 0; g < NGRP; ++g) accg[g] = sWb[g];

#pragma unroll
    for (int i = 0; i < 4; ++i) {
#pragma unroll
        for (int j = 0; j < 8; ++j) {
            const float d = p[i] * 100.0f / dimj[j];
            const float s = __sinf(d);
            const float c = __cosf(d);
            const int es = i * 16 + j;   // sin slot
            const int ec = es + 8;       // cos slot
#pragma unroll
            for (int g = 0; g < NGRP; ++g)
                accg[g] += sWg[g * EMBD + es] * s + sWg[g * EMBD + ec] * c;
        }
    }

#pragma unroll
    for (int g = 0; g < NGRP; ++g) {
        const float aw = fmaxf(accg[g], 0.0f);
        const long long idx = (((long long)g * N_ROI) + n) * M_REF + m;
        logits[idx] += __logf(aw + 1e-6f) + 1e-12f;
    }
}

// ---------------------------------------------------------------------------
// Row softmax over m=2048; one block (256 threads) per (g,n) row; bf16 output.
__global__ __launch_bounds__(256)
void softmax_kernel(const float* __restrict__ logits, bf16* __restrict__ soft) {
    const long long row = blockIdx.x;
    const float* in = logits + row * M_REF;
    bf16* out = soft + row * M_REF;
    const int tid = threadIdx.x;

    float vals[8];
    float mx = -3.0e38f;
#pragma unroll
    for (int t = 0; t < 8; ++t) {
        vals[t] = in[tid + t * 256];
        mx = fmaxf(mx, vals[t]);
    }
    __shared__ float red[256];
    red[tid] = mx;
    __syncthreads();
    for (int s = 128; s > 0; s >>= 1) {
        if (tid < s) red[tid] = fmaxf(red[tid], red[tid + s]);
        __syncthreads();
    }
    mx = red[0];
    __syncthreads();

    float sum = 0.0f;
#pragma unroll
    for (int t = 0; t < 8; ++t) {
        vals[t] = __expf(vals[t] - mx);
        sum += vals[t];
    }
    red[tid] = sum;
    __syncthreads();
    for (int s = 128; s > 0; s >>= 1) {
        if (tid < s) red[tid] += red[tid + s];
        __syncthreads();
    }
    const float inv = 1.0f / red[0];
#pragma unroll
    for (int t = 0; t < 8; ++t) out[tid + t * 256] = (bf16)(vals[t] * inv);
}

// ---------------------------------------------------------------------------
extern "C" void kernel_launch(void* const* d_in, const int* in_sizes, int n_in,
                              void* d_out, int out_size, void* d_ws, size_t ws_size,
                              hipStream_t stream) {
    const float* roi   = (const float*)d_in[0];   // (512,1024)
    const float* ref   = (const float*)d_in[1];   // (2048,1024)
    const float* bbox  = (const float*)d_in[2];   // (512,4)
    const float* bboxr = (const float*)d_in[3];   // (2048,4)
    const float* Wg_w  = (const float*)d_in[4];   // (16,64)
    const float* Wg_b  = (const float*)d_in[5];   // (16,)
    const float* Wq_w  = (const float*)d_in[6];   // (1024,1024)
    const float* Wq_b  = (const float*)d_in[7];   // (1024,)
    const float* Wk_w  = (const float*)d_in[8];   // (1024,1024)
    const float* Wk_b  = (const float*)d_in[9];   // (1024,)
    const float* Wv_w  = (const float*)d_in[10];  // (16,64,1024)
    const float* Wv_b  = (const float*)d_in[11];  // (1024,)
    const float* u     = (const float*)d_in[12];  // (16,1,64)
    float* out = (float*)d_out;                   // (512,1024) fp32

    // ---- workspace carve-up (256B aligned regions) ----
    uint8_t* ws = (uint8_t*)d_ws;
    size_t off = 0;
    auto carve = [&](size_t bytes) {
        void* p = ws + off;
        off = (off + bytes + 255) & ~(size_t)255;
        return p;
    };
    bf16*  roi_bf = (bf16*)carve((size_t)N_ROI * FEATD * 2);
    bf16*  ref_bf = (bf16*)carve((size_t)M_REF * FEATD * 2);
    bf16*  wq_bf  = (bf16*)carve((size_t)FEATD * FEATD * 2);
    bf16*  wk_bf  = (bf16*)carve((size_t)FEATD * FEATD * 2);
    bf16*  wv_bf  = (bf16*)carve((size_t)NGRP * DGRP * FEATD * 2);
    bf16*  Qb     = (bf16*)carve((size_t)N_ROI * FEATD * 2);
    bf16*  Kb     = (bf16*)carve((size_t)M_REF * FEATD * 2);
    bf16*  Pb     = (bf16*)carve((size_t)NGRP * DGRP * M_REF * 2);   // [g][o][m]
    float* affc   = (float*)carve((size_t)NGRP * M_REF * 4);
    float* logits = (float*)carve((size_t)NGRP * N_ROI * M_REF * 4); // [g][n][m]
    bf16*  soft   = (bf16*)carve((size_t)NGRP * N_ROI * M_REF * 2);  // [g][n][m]

    auto cvt = [&](const float* src, bf16* dst, long long n) {
        int blocks = (int)((n + 255) / 256);
        cvt_bf16_kernel<<<blocks, 256, 0, stream>>>(src, dst, n);
    };
    cvt(roi,  roi_bf, (long long)N_ROI * FEATD);
    cvt(ref,  ref_bf, (long long)M_REF * FEATD);
    cvt(Wq_w, wq_bf,  (long long)FEATD * FEATD);
    cvt(Wk_w, wk_bf,  (long long)FEATD * FEATD);
    cvt(Wv_w, wv_bf,  (long long)NGRP * DGRP * FEATD);

    // Q = roi @ Wq^T + bq   (bf16 out); N=1024 = 4 blocks * 4 waves * 64
    gemm_tn_kernel<4, true><<<dim3(FEATD / 256, N_ROI / 16, 1), 128, 0, stream>>>(
        roi_bf, wq_bf, Qb, FEATD, FEATD, FEATD, FEATD,
        0, 0, 0, Wq_b, 0, 1.0f);

    // K = ref @ Wk^T + bk   (bf16 out)
    gemm_tn_kernel<4, true><<<dim3(FEATD / 256, M_REF / 16, 1), 128, 0, stream>>>(
        ref_bf, wk_bf, Kb, FEATD, FEATD, FEATD, FEATD,
        0, 0, 0, Wk_b, 0, 1.0f);

    // P[g][o][m] = sum_f Wv[g,o,f] * ref[m,f]   (bf16 out, batched over g)
    gemm_tn_kernel<4, true><<<dim3(M_REF / 256, DGRP / 16, NGRP), 128, 0, stream>>>(
        wv_bf, ref_bf, Pb, FEATD, FEATD, FEATD, M_REF,
        (long long)DGRP * FEATD, 0, (long long)DGRP * M_REF, nullptr, 0, 1.0f);

    // aff_c[g][m] = u[g] . K[g][m]
    affc_kernel<<<(NGRP * M_REF) / 256, 256, 0, stream>>>(Kb, u, affc);

    // logits[g][n][m] = (Q_g K_g^T + aff_c) / sqrt(64)   (fp32 out, batched)
    gemm_tn_kernel<4, false><<<dim3(M_REF / 256, N_ROI / 16, NGRP), 128, 0, stream>>>(
        Qb, Kb, logits, DGRP, FEATD, FEATD, M_REF,
        (long long)DGRP, (long long)DGRP, (long long)N_ROI * M_REF,
        affc, (long long)M_REF, 0.125f);

    // logits += log(relu(Wg . emb(n,m)) + 1e-6) + 1e-12
    posw_kernel<<<dim3(M_REF / 256, N_ROI, 1), 256, 0, stream>>>(
        bbox, bboxr, Wg_w, Wg_b, logits);

    // softmax over m  ->  bf16
    softmax_kernel<<<NGRP * N_ROI, 256, 0, stream>>>(logits, soft);

    // out[n, g*64+o] = sum_m soft[g][n][m] * P[g][o][m] + Wv_b[g*64+o]  (fp32 out)
    // N=64 -> NT=1: 1 block-x * 4 waves * 16 cols = 64, exact.
    gemm_tn_kernel<1, false><<<dim3(1, N_ROI / 16, NGRP), 128, 0, stream>>>(
        soft, Pb, out, M_REF, M_REF, M_REF, FEATD,
        (long long)N_ROI * M_REF, (long long)DGRP * M_REF, (long long)DGRP,
        Wv_b, (long long)DGRP, 1.0f);
}
